// BinaryNet2_71854802862511
// MI455X (gfx1250) — 8-bit matrix operand order repaired and checked by reading, compiled, not run on hardware
//
#include <hip/hip_runtime.h>
#include <stdint.h>


typedef __attribute__((ext_vector_type(16))) int   v16i;
typedef __attribute__((ext_vector_type(4)))  int   v4i_t;
typedef __attribute__((ext_vector_type(2)))  int   v2i_t;
typedef __attribute__((ext_vector_type(8)))  float v8f;

#define FP8_P1 0x38u
#define FP8_M1 0xB8u

#define GM 8192
#define GN 4096
#define GK 4096

__device__ __forceinline__ unsigned int enc_sign(float v) {
  return v > 0.0f ? FP8_P1 : (v < 0.0f ? FP8_M1 : 0u);
}

__global__ __launch_bounds__(256) void sign4_kernel(const float4* __restrict__ in,
                                                    unsigned int* __restrict__ out,
                                                    int n4) {
  int i = blockIdx.x * 256 + threadIdx.x;
  if (i < n4) {
    float4 v = in[i];
    out[i] = enc_sign(v.x) | (enc_sign(v.y) << 8) |
             (enc_sign(v.z) << 16) | (enc_sign(v.w) << 24);
  }
}

__global__ __launch_bounds__(256) void w3_pad_kernel(const float* __restrict__ w3,
                                                     unsigned char* __restrict__ out) {
  int i = blockIdx.x * 256 + threadIdx.x;
  int row = i >> 12;
  unsigned char s = 0;
  if (row < 10) s = (unsigned char)enc_sign(w3[i]);
  out[i] = s;
}

__device__ __forceinline__ v16i load_a_frag(const unsigned char* p) {
  v16i f;
#pragma unroll
  for (int q = 0; q < 8; q++) {
    v2i_t w = *(const v2i_t*)(p + (q >> 1) * 32 + (q & 1) * 8);
    f[2 * q]     = w.x;
    f[2 * q + 1] = w.y;
  }
  return f;
}
__device__ __forceinline__ v16i load_b_frag(const unsigned char* p) {
  v16i f;
#pragma unroll
  for (int j = 0; j < 4; j++) {
    v4i_t w = *(const v4i_t*)(p + j * 32);
    f[4 * j]     = w.x;
    f[4 * j + 1] = w.y;
    f[4 * j + 2] = w.z;
    f[4 * j + 3] = w.w;
  }
  return f;
}

#define LDSS 144

__global__ __launch_bounds__(256) void gemm_sign_kernel(
    const unsigned char* __restrict__ A,
    const unsigned char* __restrict__ B,
    unsigned char* __restrict__ C)
{
  __shared__ __align__(16) unsigned char As[2][128 * LDSS];
  __shared__ __align__(16) unsigned char Bs[2][256 * LDSS];

  const int t    = threadIdx.x;
  const int lane = t & 31;
  const int wave = t >> 5;
  const int wm   = wave >> 2;
  const int wn   = wave & 3;
  const int lh   = lane >> 4;
  const int l16  = lane & 15;

  const int Mbase = blockIdx.y * 128;
  const int Nbase = blockIdx.x * 256;

  const int      row0  = t >> 3;
  const int      col0  = (t & 7) << 4;
  const unsigned aoff0 = (unsigned)(Mbase + row0) * GK + (unsigned)col0;
  const unsigned boff0 = (unsigned)(Nbase + row0) * GK + (unsigned)col0;
  const unsigned lds0  = (unsigned)(row0 * LDSS + col0);

  auto issue = [&](unsigned abase, unsigned bbase, unsigned kb)
      __attribute__((always_inline)) {
    const unsigned ga = aoff0 + kb;
    const unsigned gb = boff0 + kb;
#pragma unroll
    for (int s = 0; s < 4; s++)
      asm volatile("global_load_async_to_lds_b128 %0, %1, %2"
                   :: "v"(abase + lds0 + (unsigned)(s * 32 * LDSS)),
                      "v"(ga + (unsigned)(s * 32 * GK)), "s"(A) : "memory");
#pragma unroll
    for (int s = 0; s < 8; s++)
      asm volatile("global_load_async_to_lds_b128 %0, %1, %2"
                   :: "v"(bbase + lds0 + (unsigned)(s * 32 * LDSS)),
                      "v"(gb + (unsigned)(s * 32 * GK)), "s"(B) : "memory");
  };

  v8f acc[4][4] = {};

  auto compute = [&](const unsigned char* Ab, const unsigned char* Bb)
      __attribute__((always_inline)) {
    v16i af[4], bf[4];
#pragma unroll
    for (int ms = 0; ms < 4; ms++)
      af[ms] = load_a_frag(Ab + (wm * 64 + ms * 16 + l16) * LDSS + lh * 16);
#pragma unroll
    for (int ns = 0; ns < 4; ns++)
      bf[ns] = load_b_frag(Bb + (wn * 64 + ns * 16 + l16) * LDSS + lh * 16);
#pragma unroll
    for (int ms = 0; ms < 4; ms++)
#pragma unroll
      for (int ns = 0; ns < 4; ns++)
        acc[ms][ns] = __builtin_amdgcn_wmma_f32_16x16x128_fp8_fp8(
            af[ms], bf[ns], (short)0, acc[ms][ns], false, false);
  };

  const unsigned A0 = (unsigned)(uintptr_t)&As[0][0];
  const unsigned A1 = (unsigned)(uintptr_t)&As[1][0];
  const unsigned B0 = (unsigned)(uintptr_t)&Bs[0][0];
  const unsigned B1 = (unsigned)(uintptr_t)&Bs[1][0];

  constexpr int nk = GK / 128;
  issue(A0, B0, 0u);
  for (int ii = 0; ii < nk / 2 - 1; ii++) {
    const unsigned kb = (unsigned)ii * 256u;
    issue(A1, B1, kb + 128u);
    asm volatile("s_wait_asynccnt 12" ::: "memory");
    __syncthreads();
    compute(&As[0][0], &Bs[0][0]);
    __syncthreads();
    issue(A0, B0, kb + 256u);
    asm volatile("s_wait_asynccnt 12" ::: "memory");
    __syncthreads();
    compute(&As[1][0], &Bs[1][0]);
    __syncthreads();
  }
  issue(A1, B1, (unsigned)(nk - 1) * 128u);
  asm volatile("s_wait_asynccnt 12" ::: "memory");
  __syncthreads();
  compute(&As[0][0], &Bs[0][0]);
  __syncthreads();
  asm volatile("s_wait_asynccnt 0" ::: "memory");
  __syncthreads();
  compute(&As[1][0], &Bs[1][0]);

#pragma unroll
  for (int ms = 0; ms < 4; ms++) {
#pragma unroll
    for (int ns = 0; ns < 4; ns++) {
      int n = Nbase + wn * 64 + ns * 16 + l16;
#pragma unroll
      for (int r = 0; r < 8; r++) {
        int m = Mbase + wm * 64 + ms * 16 + r + 8 * lh;
        C[(size_t)m * GN + n] = (unsigned char)enc_sign(acc[ms][ns][r]);
      }
    }
  }
}

__global__ __launch_bounds__(256) void final_sign_kernel(
    const unsigned char* __restrict__ A,
    const unsigned char* __restrict__ B,
    float* __restrict__ out)
{
  const int t    = threadIdx.x;
  const int lane = t & 31;
  const int wave = t >> 5;
  const int lh   = lane >> 4;
  const int l16  = lane & 15;
  const int mrow = blockIdx.x * 128 + wave * 16 + l16;

  v8f acc = {};
  const unsigned char* pa = A + (size_t)mrow * GK + lh * 16;
  const unsigned char* pb = B + (size_t)l16 * GK + lh * 16;
  for (int kb = 0; kb < GK; kb += 128) {
    v16i a = load_a_frag(pa + kb);
    v16i b = load_b_frag(pb + kb);
    acc = __builtin_amdgcn_wmma_f32_16x16x128_fp8_fp8(a, b, (short)0, acc,
                                                      false, false);
  }
  if (l16 < 10) {
#pragma unroll
    for (int r = 0; r < 8; r++) {
      int m   = blockIdx.x * 128 + wave * 16 + r + 8 * lh;
      float v = acc[r];
      out[(size_t)m * 10 + l16] = v > 0.0f ? 1.0f : (v < 0.0f ? -1.0f : 0.0f);
    }
  }
}

extern "C" void kernel_launch(void* const* d_in, const int* in_sizes, int n_in,
                              void* d_out, int out_size, void* d_ws, size_t ws_size,
                              hipStream_t stream) {
  const float* x  = (const float*)d_in[0];
  const float* W0 = (const float*)d_in[1];
  const float* W1 = (const float*)d_in[2];
  const float* W2 = (const float*)d_in[3];
  const float* W3 = (const float*)d_in[4];
  float* out = (float*)d_out;

  unsigned char* ws   = (unsigned char*)d_ws;
  unsigned char* actA = ws;
  unsigned char* actB = actA + (size_t)GM * GK;
  unsigned char* w0f  = actB + (size_t)GM * GN;
  unsigned char* w1f  = w0f + (size_t)GN * GK;
  unsigned char* w2f  = w1f + (size_t)GN * GK;
  unsigned char* w3f  = w2f + (size_t)GN * GK;

  {
    int n4 = GM * GK / 4;
    sign4_kernel<<<n4 / 256, 256, 0, stream>>>((const float4*)x,
                                               (unsigned int*)actA, n4);
  }
  {
    int n4 = GN * GK / 4;
    sign4_kernel<<<n4 / 256, 256, 0, stream>>>((const float4*)W0,
                                               (unsigned int*)w0f, n4);
    sign4_kernel<<<n4 / 256, 256, 0, stream>>>((const float4*)W1,
                                               (unsigned int*)w1f, n4);
    sign4_kernel<<<n4 / 256, 256, 0, stream>>>((const float4*)W2,
                                               (unsigned int*)w2f, n4);
  }
  w3_pad_kernel<<<(16 * GK) / 256, 256, 0, stream>>>(W3, w3f);

  dim3 grid(GN / 256, GM / 128);
  gemm_sign_kernel<<<grid, 256, 0, stream>>>(actA, w0f, actB);
  gemm_sign_kernel<<<grid, 256, 0, stream>>>(actB, w1f, actA);
  gemm_sign_kernel<<<grid, 256, 0, stream>>>(actA, w2f, actB);
  final_sign_kernel<<<GM / 128, 256, 0, stream>>>(actB, w3f, out);
}
